// VanillaMultiHeadAttention_61375082660557
// MI455X (gfx1250) — matrix operand order repaired and checked by reading, not compiled, not run on hardware
//
#include <hip/hip_runtime.h>
#include <hip/hip_bf16.h>
#include <stdint.h>

// Problem sizes (B=1)
#define LSEQ   2048
#define DMODEL 768
#define NHEAD  12
#define DHEAD  64
#define D3     2304   // 3*DMODEL

typedef __attribute__((ext_vector_type(16))) __bf16 v16bf;
typedef __attribute__((ext_vector_type(8)))  float  v8f;

union Frag {
  v16bf v;
  unsigned int u[8];
};

__device__ __forceinline__ unsigned short f32_to_bf16(float f) {
  unsigned int u = __float_as_uint(f);
  u = u + 0x7FFFu + ((u >> 16) & 1u);   // round-to-nearest-even
  return (unsigned short)(u >> 16);
}

// A-matrix (16x32 bf16) K index for VGPR v, lane-half sel (contiguous: k = 16*sel + 2*v, matches B)
__device__ __forceinline__ int kidxA(int v, int sel) {
  return sel * 16 + 2 * v;
}

__device__ __forceinline__ v8f wmma_bf16(Frag a, Frag b, v8f c) {
  return __builtin_amdgcn_wmma_f32_16x16x32_bf16(
      /*neg_a=*/false, a.v, /*neg_b=*/false, b.v,
      /*c_mod=*/(short)0, c, /*reuse_a=*/false, /*reuse_b=*/false);
}

// blockDim.x == 256 assumed
__device__ __forceinline__ float block_reduce_sum(float v, float* buf) {
  const int t = threadIdx.x;
  buf[t] = v;
  __syncthreads();
  for (int s = 128; s > 0; s >>= 1) {
    if (t < s) buf[t] += buf[t + s];
    __syncthreads();
  }
  float r = buf[0];
  __syncthreads();
  return r;
}

// ---------------------------------------------------------------------------
// 1) LayerNorm(x) -> bf16 activations  (one block per token row)
// ---------------------------------------------------------------------------
__global__ void ln_x_kernel(const float* __restrict__ x,
                            const float* __restrict__ w,
                            const float* __restrict__ b,
                            unsigned short* __restrict__ out) {
  __shared__ float buf[256];
  const int row = blockIdx.x;
  const int t = threadIdx.x;
  const float* xr = x + (size_t)row * DMODEL;
  float vals[3];
  float s = 0.f, ss = 0.f;
#pragma unroll
  for (int i = 0; i < 3; i++) {
    float v = xr[i * 256 + t];
    vals[i] = v;
    s += v;
    ss += v * v;
  }
  s = block_reduce_sum(s, buf);
  ss = block_reduce_sum(ss, buf);
  const float inv_d = 1.0f / (float)DMODEL;
  float mean = s * inv_d;
  float var = ss * inv_d - mean * mean;       // population variance
  float inv = rsqrtf(var + 1e-5f);
#pragma unroll
  for (int i = 0; i < 3; i++) {
    int c = i * 256 + t;
    float y = (vals[i] - mean) * inv * w[c] + b[c];
    out[(size_t)row * DMODEL + c] = f32_to_bf16(y);
  }
}

// ---------------------------------------------------------------------------
// 2) fp32 weight [K][N] -> bf16 transposed [N][K] (K contiguous for B-frags)
// ---------------------------------------------------------------------------
__global__ void convert_transpose_kernel(const float* __restrict__ w,
                                         unsigned short* __restrict__ wT,
                                         int K, int N) {
  size_t idx = (size_t)blockIdx.x * blockDim.x + threadIdx.x;
  size_t total = (size_t)K * N;
  if (idx >= total) return;
  int k = (int)(idx / N);
  int n = (int)(idx % N);
  wT[(size_t)n * K + k] = f32_to_bf16(w[idx]);
}

// ---------------------------------------------------------------------------
// 3) bf16 GEMM: C[M][N] f32 = A[M][K] bf16 x BT[N][K] bf16
//    one wave -> 16x64 output strip, 4 WMMA accumulators, K-steps of 32
// ---------------------------------------------------------------------------
__global__ void gemm_bf16_tn(const unsigned short* __restrict__ A,
                             const unsigned short* __restrict__ BT,
                             float* __restrict__ C,
                             int M, int N, int K) {
  const int wid = (int)((blockIdx.x * blockDim.x + threadIdx.x) >> 5);
  const int lane = threadIdx.x & 31;
  const int nstrips = N >> 6;
  const int total = (M >> 4) * nstrips;
  if (wid >= total) return;
  const int mt = wid / nstrips;
  const int ns = wid % nstrips;
  const int nI = lane & 15;
  const int sel = lane >> 4;
  const int rowA = (mt << 4) + nI;   // A-layout: lane%16 = M row

  v8f acc0 = {}, acc1 = {}, acc2 = {}, acc3 = {};

  for (int k0 = 0; k0 < K; k0 += 32) {
    Frag a;
#pragma unroll
    for (int v = 0; v < 8; v++) {
      a.u[v] = *reinterpret_cast<const unsigned int*>(
          A + (size_t)rowA * K + k0 + kidxA(v, sel));
    }
#pragma unroll
    for (int f = 0; f < 4; f++) {
      const int ncol = (ns << 6) + (f << 4) + nI;  // B-layout: lane%16 = N col
      Frag b;
#pragma unroll
      for (int v = 0; v < 8; v++) {
        b.u[v] = *reinterpret_cast<const unsigned int*>(
            BT + (size_t)ncol * K + k0 + (sel << 4) + (v << 1));
      }
      if (f == 0) acc0 = wmma_bf16(a, b, acc0);
      else if (f == 1) acc1 = wmma_bf16(a, b, acc1);
      else if (f == 2) acc2 = wmma_bf16(a, b, acc2);
      else acc3 = wmma_bf16(a, b, acc3);
    }
  }

  v8f accs[4] = {acc0, acc1, acc2, acc3};
#pragma unroll
  for (int f = 0; f < 4; f++) {
    const int ncol = (ns << 6) + (f << 4) + nI;
#pragma unroll
    for (int g = 0; g < 8; g++) {
      const int r = (mt << 4) + g + (sel << 3);   // C-layout row
      C[(size_t)r * N + ncol] = accs[f][g];
    }
  }
}

// ---------------------------------------------------------------------------
// 4) QK-LayerNorm (over full D, no bias) + RoPE; scatter to WMMA-friendly
//    layouts: Q,K -> [H][L][DH] bf16 ; V -> [H][DH][L] bf16 (transposed)
// ---------------------------------------------------------------------------
__global__ void qk_ln_rope_kernel(const float* __restrict__ qkv,
                                  const float* __restrict__ qw,
                                  const float* __restrict__ kw,
                                  unsigned short* __restrict__ qo,
                                  unsigned short* __restrict__ ko,
                                  unsigned short* __restrict__ vT) {
  __shared__ float shq[DMODEL];
  __shared__ float shk[DMODEL];
  __shared__ float buf[256];
  const int pos = blockIdx.x;
  const int t = threadIdx.x;
  const float* row = qkv + (size_t)pos * D3;

  float qv[3], kvv[3];
  float qs = 0.f, qss = 0.f, ks = 0.f, kss = 0.f;
#pragma unroll
  for (int i = 0; i < 3; i++) {
    int c = i * 256 + t;
    float q = row[c];
    float k = row[DMODEL + c];
    qv[i] = q; kvv[i] = k;
    qs += q; qss += q * q;
    ks += k; kss += k * k;
  }
  qs = block_reduce_sum(qs, buf);
  qss = block_reduce_sum(qss, buf);
  ks = block_reduce_sum(ks, buf);
  kss = block_reduce_sum(kss, buf);
  const float inv_d = 1.0f / (float)DMODEL;
  float qm = qs * inv_d, km = ks * inv_d;
  float qr = rsqrtf(qss * inv_d - qm * qm + 1e-5f);
  float kr = rsqrtf(kss * inv_d - km * km + 1e-5f);
#pragma unroll
  for (int i = 0; i < 3; i++) {
    int c = i * 256 + t;
    shq[c] = (qv[i] - qm) * qr * qw[c];
    shk[c] = (kvv[i] - km) * kr * kw[c];
  }
  __syncthreads();

  const float kLog = 9.210340371976184f / 32.0f;   // ln(10000)/(DH/2)
#pragma unroll
  for (int i = 0; i < 3; i++) {
    int c = i * 256 + t;
    int h = c >> 6, dh = c & 63, j = dh & 31;
    float inv_freq = __expf(-(float)j * kLog);     // 10000^(-j/32)
    float ang = (float)pos * inv_freq;
    float ca = __cosf(ang), sa = __sinf(ang);
    int base = h << 6;
    float q1 = shq[base + j], q2 = shq[base + 32 + j];
    float k1 = shk[base + j], k2 = shk[base + 32 + j];
    float qr_v = (dh < 32) ? (q1 * ca - q2 * sa) : (q2 * ca + q1 * sa);
    float kr_v = (dh < 32) ? (k1 * ca - k2 * sa) : (k2 * ca + k1 * sa);
    size_t oidx = ((size_t)(h * LSEQ + pos)) * DHEAD + dh;
    qo[oidx] = f32_to_bf16(qr_v);
    ko[oidx] = f32_to_bf16(kr_v);
    vT[((size_t)(h * DHEAD + dh)) * LSEQ + pos] = f32_to_bf16(row[2 * DMODEL + c]);
  }
}

// ---------------------------------------------------------------------------
// 5) Flash-style attention: one wave per (head, 16-row Q tile).
//    kv tiles of 32 so P (16x32, two C-frags) becomes exactly one bf16 A-frag.
//    scores = QK^T * 0.125 + bias(same_doc & mask), online softmax, O += P*V.
// ---------------------------------------------------------------------------
__global__ void attn_kernel(const unsigned short* __restrict__ Q,   // [H][L][DH]
                            const unsigned short* __restrict__ Kb,  // [H][L][DH]
                            const unsigned short* __restrict__ VT,  // [H][DH][L]
                            const int* __restrict__ seq_id,
                            const unsigned char* __restrict__ amask,
                            unsigned short* __restrict__ ctx) {     // [L][D] bf16
  __shared__ float pbuf[8][16 * 33];   // per-wave P staging, padded stride 33
  const int wslot = threadIdx.x >> 5;
  const int wid = (int)((blockIdx.x * blockDim.x + threadIdx.x) >> 5);
  const int lane = threadIdx.x & 31;
  const int head = wid >> 7;           // / (L/16)
  const int qt = wid & 127;
  const int qbase = qt << 4;
  const int nI = lane & 15;
  const int sel = lane >> 4;

  // Q A-fragments for dh halves [0,32) and [32,64)
  Frag aq[2];
  const unsigned short* qrp = Q + ((size_t)(head * LSEQ + qbase + nI)) * DHEAD;
#pragma unroll
  for (int hb = 0; hb < 2; hb++)
#pragma unroll
    for (int v = 0; v < 8; v++)
      aq[hb].u[v] = *reinterpret_cast<const unsigned int*>(
          qrp + hb * 32 + kidxA(v, sel));

  // per-lane row state: this lane's 8 C-layout rows are qbase + g + 8*sel
  float mstate[8], lstate[8];
  int sq[8], mq[8];
#pragma unroll
  for (int g = 0; g < 8; g++) {
    int r = qbase + g + (sel << 3);
    sq[g] = seq_id[r];
    mq[g] = (amask[r] != 0);
    mstate[g] = -1e30f;
    lstate[g] = 0.f;
  }
  v8f o0 = {}, o1 = {}, o2 = {}, o3 = {};
  const float scale = 0.125f;          // 1/sqrt(64)

  for (int kt = 0; kt < (LSEQ / 32); kt++) {
    const int kvbase = kt << 5;

    // S (16x32) = Q(16x64) x K^T(64x32): two N-halves, two dh-halves each
    v8f s0 = {}, s1 = {};
#pragma unroll
    for (int hb = 0; hb < 2; hb++) {
      const unsigned short* kp0 =
          Kb + ((size_t)(head * LSEQ + kvbase + nI)) * DHEAD + hb * 32 + (sel << 4);
      const unsigned short* kp1 = kp0 + (size_t)16 * DHEAD;
      Frag b0, b1;
#pragma unroll
      for (int v = 0; v < 8; v++) {
        b0.u[v] = *reinterpret_cast<const unsigned int*>(kp0 + (v << 1));
        b1.u[v] = *reinterpret_cast<const unsigned int*>(kp1 + (v << 1));
      }
      s0 = wmma_bf16(aq[hb], b0, s0);
      s1 = wmma_bf16(aq[hb], b1, s1);
    }

    // bias + online softmax (row reductions across the 16 lanes of each half)
    const int c0 = kvbase + nI, c1 = c0 + 16;
    const int sk0 = seq_id[c0], sk1 = seq_id[c1];
    const int mk0 = (amask[c0] != 0), mk1 = (amask[c1] != 0);
    float p0[8], p1[8], aold[8];
#pragma unroll
    for (int g = 0; g < 8; g++) {
      float b0 = (sq[g] == sk0 && mq[g] && mk0) ? 1.f : 0.f;
      float b1 = (sq[g] == sk1 && mq[g] && mk1) ? 1.f : 0.f;
      float v0 = s0[g] * scale + b0;
      float v1 = s1[g] * scale + b1;
      float tm = fmaxf(v0, v1);
#pragma unroll
      for (int off = 1; off < 16; off <<= 1)
        tm = fmaxf(tm, __shfl_xor(tm, off, 16));
      float nm = fmaxf(mstate[g], tm);
      float al = __expf(mstate[g] - nm);
      float e0 = __expf(v0 - nm);
      float e1 = __expf(v1 - nm);
      float rs = e0 + e1;
#pragma unroll
      for (int off = 1; off < 16; off <<= 1)
        rs += __shfl_xor(rs, off, 16);
      lstate[g] = lstate[g] * al + rs;
      mstate[g] = nm;
      aold[g] = al;
      p0[g] = e0;
      p1[g] = e1;
    }
#pragma unroll
    for (int g = 0; g < 8; g++) {
      o0[g] *= aold[g]; o1[g] *= aold[g]; o2[g] *= aold[g]; o3[g] *= aold[g];
    }

    // C-layout P -> A-layout bf16 via per-wave LDS tile (same-wave DS in-order)
#pragma unroll
    for (int g = 0; g < 8; g++) {
      pbuf[wslot][(g + (sel << 3)) * 33 + nI] = p0[g];
      pbuf[wslot][(g + (sel << 3)) * 33 + 16 + nI] = p1[g];
    }
    Frag pa;
#pragma unroll
    for (int v = 0; v < 8; v++) {
      int kk = kidxA(v, sel);
      float f0 = pbuf[wslot][nI * 33 + kk];
      float f1 = pbuf[wslot][nI * 33 + kk + 1];
      pa.u[v] = (unsigned int)f32_to_bf16(f0) |
                ((unsigned int)f32_to_bf16(f1) << 16);
    }

    // O(16x64) += P(16x32) x V(32x64): 4 B-fragments from transposed V
#pragma unroll
    for (int f = 0; f < 4; f++) {
      const int dh = (f << 4) + nI;
      const unsigned short* vp =
          VT + ((size_t)(head * DHEAD + dh)) * LSEQ + kvbase + (sel << 4);
      Frag bv;
#pragma unroll
      for (int v = 0; v < 8; v++)
        bv.u[v] = *reinterpret_cast<const unsigned int*>(vp + (v << 1));
      if (f == 0) o0 = wmma_bf16(pa, bv, o0);
      else if (f == 1) o1 = wmma_bf16(pa, bv, o1);
      else if (f == 2) o2 = wmma_bf16(pa, bv, o2);
      else o3 = wmma_bf16(pa, bv, o3);
    }
  }

  // epilogue: normalize by row sum, write ctx[L][H*DH] bf16
  v8f os[4] = {o0, o1, o2, o3};
#pragma unroll
  for (int f = 0; f < 4; f++) {
    const int col = head * DHEAD + (f << 4) + nI;
#pragma unroll
    for (int g = 0; g < 8; g++) {
      const int r = qbase + g + (sel << 3);
      ctx[(size_t)r * DMODEL + col] = f32_to_bf16(os[f][g] / lstate[g]);
    }
  }
}

// ---------------------------------------------------------------------------
// launch
// ---------------------------------------------------------------------------
extern "C" void kernel_launch(void* const* d_in, const int* in_sizes, int n_in,
                              void* d_out, int out_size, void* d_ws, size_t ws_size,
                              hipStream_t stream) {
  (void)in_sizes; (void)n_in; (void)out_size; (void)ws_size;

  const float* x      = (const float*)d_in[0];
  const unsigned char* amask = (const unsigned char*)d_in[1];  // bool mask
  const int* seq_id   = (const int*)d_in[2];
  const float* ln_w   = (const float*)d_in[3];
  const float* ln_b   = (const float*)d_in[4];
  const float* w_qkv  = (const float*)d_in[5];
  const float* q_ln_w = (const float*)d_in[6];
  const float* k_ln_w = (const float*)d_in[7];
  const float* w_out  = (const float*)d_in[8];
  float* out = (float*)d_out;

  char* ws = (char*)d_ws;
  size_t off = 0;
  auto alloc = [&](size_t bytes) {
    void* p = ws + off;
    off += (bytes + 255) & ~(size_t)255;
    return p;
  };
  unsigned short* h_bf   = (unsigned short*)alloc((size_t)LSEQ * DMODEL * 2);
  unsigned short* wqkvT  = (unsigned short*)alloc((size_t)DMODEL * D3 * 2);
  unsigned short* woutT  = (unsigned short*)alloc((size_t)DMODEL * DMODEL * 2);
  float*          qkv    = (float*)alloc((size_t)LSEQ * D3 * 4);
  unsigned short* q_bf   = (unsigned short*)alloc((size_t)NHEAD * LSEQ * DHEAD * 2);
  unsigned short* k_bf   = (unsigned short*)alloc((size_t)NHEAD * LSEQ * DHEAD * 2);
  unsigned short* vT_bf  = (unsigned short*)alloc((size_t)NHEAD * DHEAD * LSEQ * 2);
  unsigned short* ctx_bf = (unsigned short*)alloc((size_t)LSEQ * DMODEL * 2);

  // 1) LN(x) -> bf16
  ln_x_kernel<<<LSEQ, 256, 0, stream>>>(x, ln_w, ln_b, h_bf);

  // 2) weights -> bf16 transposed
  {
    int n = DMODEL * D3;
    convert_transpose_kernel<<<(n + 255) / 256, 256, 0, stream>>>(w_qkv, wqkvT, DMODEL, D3);
    n = DMODEL * DMODEL;
    convert_transpose_kernel<<<(n + 255) / 256, 256, 0, stream>>>(w_out, woutT, DMODEL, DMODEL);
  }

  // 3) QKV GEMM: 2048x2304x768 ; (2048/16)*(2304/64)=4608 waves / 8 per block
  gemm_bf16_tn<<<576, 256, 0, stream>>>(h_bf, wqkvT, qkv, LSEQ, D3, DMODEL);

  // 4) QK-LN + RoPE + layout scatter
  qk_ln_rope_kernel<<<LSEQ, 256, 0, stream>>>(qkv, q_ln_w, k_ln_w, q_bf, k_bf, vT_bf);

  // 5) attention: 12 heads * 128 q-tiles = 1536 waves / 8 per block
  attn_kernel<<<192, 256, 0, stream>>>(q_bf, k_bf, vT_bf, seq_id, amask, ctx_bf);

  // 6) output GEMM: 2048x768x768 ; (2048/16)*(768/64)=1536 waves / 8 per block
  gemm_bf16_tn<<<192, 256, 0, stream>>>(ctx_bf, woutT, out, LSEQ, DMODEL, DMODEL);
}
